// GINENet_62818191671466
// MI455X (gfx1250) — matrix operand order repaired and checked by reading, not compiled, not run on hardware
//
#include <hip/hip_runtime.h>
#include <hip/hip_bf16.h>

typedef __attribute__((ext_vector_type(16))) _Float16 v16h;
typedef __attribute__((ext_vector_type(8)))  float    v8f;

#define N_NODES 100000
#define N_EDGES 1000000
#define HID 64
#define N_GRAPHS 64

__device__ __forceinline__ v8f vzero8() {
  v8f v;
#pragma unroll
  for (int i = 0; i < 8; ++i) v[i] = 0.0f;
  return v;
}

__device__ __forceinline__ v8f wmma16(v16h a, v16h b, v8f c) {
  return __builtin_amdgcn_wmma_f32_16x16x32_f16(false, a, false, b, (short)0, c,
                                                false, false);
}

// ---------------------------------------------------------------------------
// Pack a row-major f32 weight W[K][N] into WMMA B-operand lane order (f16).
// Tile index = nt*KT + kt. lane<16 -> K offsets 0..7,16..23 of the 32-K tile,
// lane>=16 -> 8..15,24..31; column n = nt*16 + (lane&15). 16 f16/lane.
// ---------------------------------------------------------------------------
__global__ void pack_b_kernel(const float* __restrict__ W, int K, int N,
                              _Float16* __restrict__ out) {
  int tile = blockIdx.x;
  int lane = threadIdx.x;  // 0..31
  int KT = K >> 5;
  int nt = tile / KT, kt = tile % KT;
  int kbase = kt * 32 + ((lane >= 16) ? 8 : 0);
  int n = nt * 16 + (lane & 15);
  _Float16* o = out + ((size_t)(tile * 32 + lane)) * 16;
#pragma unroll
  for (int j = 0; j < 16; ++j) o[j] = (_Float16)W[(size_t)(kbase + j + (j >= 8 ? 8 : 0)) * N + n];
}

// ---------------------------------------------------------------------------
// Node encoder: h = relu(x @ node_w + node_b).  M=100000, K=32, N=64.
// One wave per 16-row tile; single K-tile; 4 N-tiles -> 4 WMMAs per wave.
// ---------------------------------------------------------------------------
__global__ __launch_bounds__(256) void node_enc_kernel(
    const float* __restrict__ x, const _Float16* __restrict__ nwp,
    const float* __restrict__ nb, float* __restrict__ h) {
  int wave = threadIdx.x >> 5, lane = threadIdx.x & 31;
  int half = lane >> 4, lm = lane & 15;
  int tile = blockIdx.x * 8 + wave;
  if (tile >= N_NODES / 16) return;
  int row0 = tile * 16;
  const float* xr = x + (size_t)(row0 + lm) * 32;
  v16h A;
#pragma unroll
  for (int j = 0; j < 8; ++j) A[j] = (_Float16)xr[half * 8 + j];
#pragma unroll
  for (int j = 0; j < 8; ++j) A[8 + j] = (_Float16)xr[16 + half * 8 + j];
  v8f acc[4];
#pragma unroll
  for (int nt = 0; nt < 4; ++nt) acc[nt] = vzero8();
#pragma unroll
  for (int nt = 0; nt < 4; ++nt) {
    v16h B = *(const v16h*)(nwp + ((size_t)(nt * 32 + lane)) * 16);
    acc[nt] = wmma16(A, B, acc[nt]);
  }
#pragma unroll
  for (int nt = 0; nt < 4; ++nt) {
    float bb = nb[nt * 16 + lm];
#pragma unroll
    for (int r = 0; r < 8; ++r) {
      float v = acc[nt][r] + bb;
      v = v > 0.0f ? v : 0.0f;
      h[(size_t)(row0 + r + half * 8) * HID + nt * 16 + lm] = v;
    }
  }
}

// ---------------------------------------------------------------------------
// Edge message + scatter: aggr[dst] += relu(h[src] + edge_attr@edge_w + edge_b)
// 64 edges per 256-thread block, 4 threads/edge (16 channels each).
// edge_w (16x64 f32 = 4KB) cached in LDS.
// ---------------------------------------------------------------------------
__global__ __launch_bounds__(256) void edge_msg_kernel(
    const float* __restrict__ h, const float* __restrict__ ea,
    const int* __restrict__ ei, const float* __restrict__ ew,
    const float* __restrict__ ebias, float* __restrict__ aggr, int E) {
  __shared__ float sew[16 * 64];
  __shared__ float seb[64];
  for (int i = threadIdx.x; i < 1024; i += 256) sew[i] = ew[i];
  if (threadIdx.x < 64) seb[threadIdx.x] = ebias[threadIdx.x];
  __syncthreads();
  int t = blockIdx.x * 256 + threadIdx.x;
  int e = t >> 2;
  int chunk = t & 3;
  if (e >= E) return;
  int src = ei[e];
  int dst = ei[E + e];
  float a[16];
#pragma unroll
  for (int k = 0; k < 16; ++k) a[k] = ea[(size_t)e * 16 + k];
  const float* hr = h + (size_t)src * HID + chunk * 16;
  float* ag = aggr + (size_t)dst * HID + chunk * 16;
#pragma unroll 4
  for (int cc = 0; cc < 16; ++cc) {
    int c = chunk * 16 + cc;
    float acc = seb[c];
#pragma unroll
    for (int k = 0; k < 16; ++k) acc += a[k] * sew[k * 64 + c];
    float m = hr[cc] + acc;
    m = m > 0.0f ? m : 0.0f;
    atomicAdd(ag + cc, m);
  }
}

// ---------------------------------------------------------------------------
// Fused GIN MLP: z = (1+eps)*h + aggr ; hid = relu(z@w1+b1) ; z2 = hid@w2+b2
// Writes z2 in place of aggr (disjoint 16-row tiles). Accumulates per-channel
// sum / sum-of-squares for BatchNorm (LDS reduce + 1 global atomic per chan).
// GEMM1: K=64 (2 ktiles) x N=128 (8 ntiles); GEMM2: K=128 x N=64.
// ---------------------------------------------------------------------------
__global__ __launch_bounds__(256) void gine_mlp_kernel(
    const float* __restrict__ h, float* az, const _Float16* __restrict__ w1p,
    const _Float16* __restrict__ w2p, const float* __restrict__ b1,
    const float* __restrict__ b2, const float* __restrict__ epsp,
    float* __restrict__ bnsum, float* __restrict__ bnsq) {
  __shared__ _Float16 hid[8][16 * 128];  // 4KB per wave
  __shared__ float redsum[64];
  __shared__ float redsq[64];
  const int wave = threadIdx.x >> 5;
  const int lane = threadIdx.x & 31;
  const int half = lane >> 4;
  const int lm = lane & 15;
  for (int i = threadIdx.x; i < 64; i += 256) {
    redsum[i] = 0.0f;
    redsq[i] = 0.0f;
  }
  __syncthreads();
  const int tile = blockIdx.x * 8 + wave;
  if (tile < N_NODES / 16) {
    const int row0 = tile * 16;
    const float epsv = 1.0f + epsp[0];
    // ---- GEMM1: z[16x64] @ w1[64x128] ----
    v8f acc[8];
#pragma unroll
    for (int nt = 0; nt < 8; ++nt) acc[nt] = vzero8();
#pragma unroll
    for (int kt = 0; kt < 2; ++kt) {
      const float* hr = h + (size_t)(row0 + lm) * HID + kt * 32;
      const float* ar = az + (size_t)(row0 + lm) * HID + kt * 32;
      int k0 = half * 8, k1 = 16 + half * 8;
      v16h A;
#pragma unroll
      for (int j = 0; j < 8; ++j)
        A[j] = (_Float16)(epsv * hr[k0 + j] + ar[k0 + j]);
#pragma unroll
      for (int j = 0; j < 8; ++j)
        A[8 + j] = (_Float16)(epsv * hr[k1 + j] + ar[k1 + j]);
#pragma unroll
      for (int nt = 0; nt < 8; ++nt) {
        v16h B = *(const v16h*)(w1p + ((size_t)((nt * 2 + kt) * 32 + lane)) * 16);
        acc[nt] = wmma16(A, B, acc[nt]);
      }
    }
    // ---- bias + relu -> LDS (f16, row-major 16x128) ----
    _Float16* hw = &hid[wave][0];
#pragma unroll
    for (int nt = 0; nt < 8; ++nt) {
      float bb = b1[nt * 16 + lm];
#pragma unroll
      for (int r = 0; r < 8; ++r) {
        float v = acc[nt][r] + bb;
        v = v > 0.0f ? v : 0.0f;
        hw[(r + half * 8) * 128 + nt * 16 + lm] = (_Float16)v;
      }
    }
    // ---- GEMM2: hid[16x128] @ w2[128x64] ----
    v8f acc2[4];
#pragma unroll
    for (int nt = 0; nt < 4; ++nt) acc2[nt] = vzero8();
#pragma unroll
    for (int kt = 0; kt < 4; ++kt) {
      const _Float16* rr = hw + lm * 128 + kt * 32;
      v16h A;
#pragma unroll
      for (int j = 0; j < 8; ++j) A[j] = rr[half * 8 + j];
#pragma unroll
      for (int j = 0; j < 8; ++j) A[8 + j] = rr[16 + half * 8 + j];
#pragma unroll
      for (int nt = 0; nt < 4; ++nt) {
        v16h B = *(const v16h*)(w2p + ((size_t)((nt * 4 + kt) * 32 + lane)) * 16);
        acc2[nt] = wmma16(A, B, acc2[nt]);
      }
    }
    // ---- bias, write z2 in place of aggr, BN partial sums ----
#pragma unroll
    for (int nt = 0; nt < 4; ++nt) {
      float bb = b2[nt * 16 + lm];
      float s = 0.0f, ss = 0.0f;
#pragma unroll
      for (int r = 0; r < 8; ++r) {
        float v = acc2[nt][r] + bb;
        az[(size_t)(row0 + r + half * 8) * HID + nt * 16 + lm] = v;
        s += v;
        ss += v * v;
      }
      s += __shfl_xor(s, 16, 32);
      ss += __shfl_xor(ss, 16, 32);
      if (half == 0) {
        atomicAdd(&redsum[nt * 16 + lm], s);
        atomicAdd(&redsq[nt * 16 + lm], ss);
      }
    }
  }
  __syncthreads();
  for (int i = threadIdx.x; i < 64; i += 256) {
    atomicAdd(&bnsum[i], redsum[i]);
    atomicAdd(&bnsq[i], redsq[i]);
  }
}

// ---------------------------------------------------------------------------
// BatchNorm (training stats, biased var) + residual + relu:  h = relu(bn(z)+h)
// One thread per 4 channels.
// ---------------------------------------------------------------------------
__global__ __launch_bounds__(256) void bn_res_kernel(
    const float* __restrict__ z, float* __restrict__ h,
    const float* __restrict__ bnsum, const float* __restrict__ bnsq,
    const float* __restrict__ gamma, const float* __restrict__ beta) {
  int t = blockIdx.x * 256 + threadIdx.x;
  if (t >= N_NODES * 16) return;
  int node = t >> 4;
  int cg = (t & 15) * 4;
  const float invN = 1.0f / (float)N_NODES;
  float4 zv = *(const float4*)(z + (size_t)node * HID + cg);
  float4 hv = *(const float4*)(h + (size_t)node * HID + cg);
  float zz[4] = {zv.x, zv.y, zv.z, zv.w};
  float hh[4] = {hv.x, hv.y, hv.z, hv.w};
  float o[4];
#pragma unroll
  for (int j = 0; j < 4; ++j) {
    int c = cg + j;
    float mu = bnsum[c] * invN;
    float var = bnsq[c] * invN - mu * mu;
    float is = rsqrtf(var + 1e-5f);
    float v = (zz[j] - mu) * is * gamma[c] + beta[c] + hh[j];
    o[j] = v > 0.0f ? v : 0.0f;
  }
  *(float4*)(h + (size_t)node * HID + cg) = make_float4(o[0], o[1], o[2], o[3]);
}

// ---------------------------------------------------------------------------
// Global mean pool (sum + counts via atomics)
// ---------------------------------------------------------------------------
__global__ __launch_bounds__(256) void pool_kernel(const float* __restrict__ h,
                                                   const int* __restrict__ batch,
                                                   float* __restrict__ pooled,
                                                   float* __restrict__ cnts) {
  int t = blockIdx.x * 256 + threadIdx.x;
  if (t >= N_NODES * 16) return;
  int node = t >> 4;
  int cg = (t & 15) * 4;
  int g = batch[node];
  float4 hv = *(const float4*)(h + (size_t)node * HID + cg);
  atomicAdd(&pooled[g * HID + cg + 0], hv.x);
  atomicAdd(&pooled[g * HID + cg + 1], hv.y);
  atomicAdd(&pooled[g * HID + cg + 2], hv.z);
  atomicAdd(&pooled[g * HID + cg + 3], hv.w);
  if ((t & 15) == 0) atomicAdd(&cnts[g], 1.0f);
}

// ---------------------------------------------------------------------------
// Output MLP over 64 graphs (tiny): pooled/cnt -> 64 -> relu 32 -> 10
// ---------------------------------------------------------------------------
__global__ void out_mlp_kernel(const float* __restrict__ pooled,
                               const float* __restrict__ cnts,
                               const float* __restrict__ w1,
                               const float* __restrict__ b1,
                               const float* __restrict__ w2,
                               const float* __restrict__ b2,
                               float* __restrict__ out) {
  int g = threadIdx.x;
  if (g >= N_GRAPHS) return;
  float c = cnts[g];
  c = c > 1.0f ? c : 1.0f;
  float ic = 1.0f / c;
  float p[64];
#pragma unroll
  for (int k = 0; k < 64; ++k) p[k] = pooled[g * HID + k] * ic;
  float hd[32];
#pragma unroll 4
  for (int o = 0; o < 32; ++o) {
    float a = b1[o];
    for (int k = 0; k < 64; ++k) a += p[k] * w1[k * 32 + o];
    hd[o] = a > 0.0f ? a : 0.0f;
  }
#pragma unroll
  for (int o = 0; o < 10; ++o) {
    float a = b2[o];
    for (int k = 0; k < 32; ++k) a += hd[k] * w2[k * 10 + o];
    out[g * 10 + o] = a;
  }
}

extern "C" void kernel_launch(void* const* d_in, const int* in_sizes, int n_in,
                              void* d_out, int out_size, void* d_ws,
                              size_t ws_size, hipStream_t stream) {
  const float* x = (const float*)d_in[0];
  const float* edge_attr = (const float*)d_in[1];
  const int* edge_index = (const int*)d_in[2];
  const int* batch = (const int*)d_in[3];
  const float* node_w = (const float*)d_in[4];
  const float* node_b = (const float*)d_in[5];
  const float* edge_w = (const float*)d_in[6];
  const float* edge_b = (const float*)d_in[7];
  const float* eps = (const float*)d_in[8];
  const float* mlp_w1 = (const float*)d_in[9];
  const float* mlp_b1 = (const float*)d_in[10];
  const float* mlp_w2 = (const float*)d_in[11];
  const float* mlp_b2 = (const float*)d_in[12];
  const float* bn_gamma = (const float*)d_in[13];
  const float* bn_beta = (const float*)d_in[14];
  const float* out_w1 = (const float*)d_in[15];
  const float* out_b1 = (const float*)d_in[16];
  const float* out_w2 = (const float*)d_in[17];
  const float* out_b2 = (const float*)d_in[18];
  float* out = (float*)d_out;

  // workspace layout
  float* h = (float*)d_ws;                    // 100000*64
  float* az = h + (size_t)N_NODES * HID;      // 100000*64 (aggr, then z2)
  float* bnsum = az + (size_t)N_NODES * HID;  // 64
  float* bnsq = bnsum + 64;                   // 64
  float* pooled = bnsq + 64;                  // 64*64
  float* cnts = pooled + 64 * 64;             // 64
  _Float16* nwp = (_Float16*)(cnts + 64);     // 4*32*16 = 2048 f16
  _Float16* w1p = nwp + 2048;                 // 4 layers * 16*32*16
  _Float16* w2p = w1p + 4 * 8192;             // 4 layers * 16*32*16

  // pre-swizzle weights into WMMA B-operand layout (f16)
  pack_b_kernel<<<4, 32, 0, stream>>>(node_w, 32, 64, nwp);
  for (int l = 0; l < 4; ++l) {
    pack_b_kernel<<<16, 32, 0, stream>>>(mlp_w1 + (size_t)l * 64 * 128, 64, 128,
                                         w1p + (size_t)l * 8192);
    pack_b_kernel<<<16, 32, 0, stream>>>(mlp_w2 + (size_t)l * 128 * 64, 128, 64,
                                         w2p + (size_t)l * 8192);
  }

  // node encoder
  node_enc_kernel<<<782, 256, 0, stream>>>(x, nwp, node_b, h);

  // 4 GINE layers
  for (int l = 0; l < 4; ++l) {
    hipMemsetAsync(az, 0, (size_t)N_NODES * HID * sizeof(float), stream);
    hipMemsetAsync(bnsum, 0, 128 * sizeof(float), stream);
    edge_msg_kernel<<<N_EDGES / 64, 256, 0, stream>>>(
        h, edge_attr, edge_index, edge_w, edge_b, az, N_EDGES);
    gine_mlp_kernel<<<782, 256, 0, stream>>>(
        h, az, w1p + (size_t)l * 8192, w2p + (size_t)l * 8192,
        mlp_b1 + (size_t)l * 128, mlp_b2 + (size_t)l * 64, eps + l, bnsum, bnsq);
    bn_res_kernel<<<N_NODES * 16 / 256, 256, 0, stream>>>(
        az, h, bnsum, bnsq, bn_gamma + (size_t)l * 64, bn_beta + (size_t)l * 64);
  }

  // pooling + output MLP
  hipMemsetAsync(pooled, 0, (64 * 64 + 64) * sizeof(float), stream);
  pool_kernel<<<N_NODES * 16 / 256, 256, 0, stream>>>(h, batch, pooled, cnts);
  out_mlp_kernel<<<1, 64, 0, stream>>>(pooled, cnts, out_w1, out_b1, out_w2,
                                       out_b2, out);
}
